// TRANSRGAT_80925773791325
// MI455X (gfx1250) — matrix operand order repaired and checked by reading, not compiled, not run on hardware
//
#include <hip/hip_runtime.h>
#include <hip/hip_bf16.h>
#include <math.h>

// Problem constants (from reference)
#define NNODES 10000
#define NEDGES 160000
#define CH     512
#define NH     4
#define NREL   8
#define NLAY   2
#define EDIM   768
#define HID    (2*CH)
#define NEG_SLOPE 0.2f
#define LN_EPS 1e-5f

typedef __attribute__((ext_vector_type(16))) __bf16 bf16x16;
typedef __attribute__((ext_vector_type(8)))  float  f32x8;

union Frag { bf16x16 v; uint4 q[2]; };

// ---------------------------------------------------------------------------
// WMMA GEMM: C[M x Ncols] = A[M x K](bf16,row-major) * B^T[Ncols x K](bf16)
// grid.z batches B/C (relations). mode: 0 = store f32
//                                       1 = bf16(relu(acc+bias))
//                                       2 = f32(acc+bias+resid)
// Requires: Ncols % 128 == 0, K % 32 == 0 (true for all uses here).
// A-tile rows >= M are clamped to row M-1 (branch-free staging); they only
// feed accumulator rows that the guarded epilogue never stores.
// ---------------------------------------------------------------------------
#define BM 128
#define BN 128
#define BK 32
#define LDT 40   // padded LDS row stride (bf16 elements)

__global__ __launch_bounds__(256)
void gemm_bf16_wmma(const __bf16* __restrict__ A,
                    const __bf16* __restrict__ B,
                    int M, int K, int Ncols,
                    long strideB, long strideC,
                    const float* __restrict__ bias,
                    const float* __restrict__ resid,
                    float* __restrict__ outF,
                    __bf16* __restrict__ outB,
                    int mode)
{
    __shared__ __bf16 As[BM * LDT];
    __shared__ __bf16 Bs[BN * LDT];

    const int tid  = threadIdx.x;
    const int bm   = blockIdx.y * BM;
    const int bn   = blockIdx.x * BN;
    const int z    = blockIdx.z;
    B += (long)z * strideB;
    float* outFz = outF ? (outF + (long)z * strideC) : (float*)0;

    const int wave = tid >> 5;
    const int lane = tid & 31;
    const int wm   = (wave & 3) * 32;   // 4 waves along M
    const int wn   = (wave >> 2) * 64;  // 2 waves along N
    const int l15  = lane & 15;
    const int kh   = lane >> 4;

    // Branch-free staging addresses: each thread owns two 16B chunks of A and
    // two of B per K-step (rows r and r+64 of the 128-row tile).
    const int row = tid >> 2;
    const int chn = tid & 3;
    int gra0 = bm + row;       if (gra0 >= M) gra0 = M - 1;
    int gra1 = bm + row + 64;  if (gra1 >= M) gra1 = M - 1;
    const __bf16* pa0 = A + (size_t)gra0 * K + chn * 8;
    const __bf16* pa1 = A + (size_t)gra1 * K + chn * 8;
    const __bf16* pb0 = B + (size_t)(bn + row) * K + chn * 8;
    const __bf16* pb1 = B + (size_t)(bn + row + 64) * K + chn * 8;
    __bf16* sa0 = &As[row * LDT + chn * 8];
    __bf16* sa1 = &As[(row + 64) * LDT + chn * 8];
    __bf16* sb0 = &Bs[row * LDT + chn * 8];
    __bf16* sb1 = &Bs[(row + 64) * LDT + chn * 8];

    f32x8 acc[2][4];
#pragma unroll
    for (int mi = 0; mi < 2; ++mi)
#pragma unroll
        for (int ni = 0; ni < 4; ++ni)
            acc[mi][ni] = (f32x8){0.f,0.f,0.f,0.f,0.f,0.f,0.f,0.f};

    for (int k0 = 0; k0 < K; k0 += BK) {
        // Issue global loads before the barrier so they overlap the previous
        // iteration's WMMAs.
        uint4 va0 = *(const uint4*)(pa0 + k0);
        uint4 va1 = *(const uint4*)(pa1 + k0);
        uint4 vb0 = *(const uint4*)(pb0 + k0);
        uint4 vb1 = *(const uint4*)(pb1 + k0);
        if (k0 + BK < K) {   // gfx1250: lowers to global_prefetch_b8
            __builtin_prefetch(pa0 + k0 + BK, 0, 1);
            __builtin_prefetch(pb0 + k0 + BK, 0, 1);
        }
        __syncthreads();     // previous iteration done reading LDS
        *(uint4*)sa0 = va0;
        *(uint4*)sa1 = va1;
        *(uint4*)sb0 = vb0;
        *(uint4*)sb1 = vb1;
        __syncthreads();

        // A frag: lanes<16 hold K in {0..7}U{16..23}, lanes>=16 {8..15}U{24..31}
        Frag a[2], b[4];
#pragma unroll
        for (int mi = 0; mi < 2; ++mi) {
            int m = wm + mi * 16 + l15;
            a[mi].q[0] = *(const uint4*)(&As[m * LDT + kh * 8]);
            a[mi].q[1] = *(const uint4*)(&As[m * LDT + 16 + kh * 8]);
        }
        // B frag: lanes<16 hold K {0..7}U{16..23}, lanes>=16 {8..15}U{24..31} (lane = column)
#pragma unroll
        for (int ni = 0; ni < 4; ++ni) {
            int n = wn + ni * 16 + l15;
            b[ni].q[0] = *(const uint4*)(&Bs[n * LDT + kh * 8]);
            b[ni].q[1] = *(const uint4*)(&Bs[n * LDT + 16 + kh * 8]);
        }
#pragma unroll
        for (int mi = 0; mi < 2; ++mi)
#pragma unroll
            for (int ni = 0; ni < 4; ++ni)
                acc[mi][ni] = __builtin_amdgcn_wmma_f32_16x16x32_bf16(
                    false, a[mi].v, false, b[ni].v,
                    (short)0, acc[mi][ni], false, false);
    }

    // Epilogue. C/D layout: VGPR v -> row v + 8*(lane>=16), col = lane&15
    const int rofs = kh * 8;
#pragma unroll
    for (int mi = 0; mi < 2; ++mi)
#pragma unroll
        for (int ni = 0; ni < 4; ++ni)
#pragma unroll
            for (int v = 0; v < 8; ++v) {
                int gr = bm + wm + mi * 16 + v + rofs;
                int gc = bn + wn + ni * 16 + l15;
                if (gr < M) {
                    float x = acc[mi][ni][v];
                    size_t idx = (size_t)gr * Ncols + gc;
                    if (mode == 0)      outFz[idx] = x;
                    else if (mode == 1) outB[idx] = (__bf16)fmaxf(x + bias[gc], 0.f);
                    else                outFz[idx] = x + bias[gc] + resid[idx];
                }
            }
}

// ---------------------------------------------------------------------------
// Elementwise / conversion helpers
// ---------------------------------------------------------------------------
__global__ void f32_to_bf16_kernel(const float* __restrict__ in,
                                   __bf16* __restrict__ out, int n) {
    int i = blockIdx.x * 256 + threadIdx.x;
    if (i < n) out[i] = (__bf16)in[i];
}

// out[b][c][r] = (bf16) in[b][r][c]
__global__ void transpose_bf16_kernel(const float* __restrict__ in,
                                      __bf16* __restrict__ out,
                                      int B, int R, int Cc) {
    long i = (long)blockIdx.x * 256 + threadIdx.x;
    long total = (long)B * R * Cc;
    if (i < total) {
        int c = (int)(i % Cc);
        long t = i / Cc;
        int r = (int)(t % R);
        int b = (int)(t / R);
        out[(long)b * R * Cc + (long)c * R + r] = (__bf16)in[i];
    }
}

__global__ void init_layer_kernel(float* __restrict__ agg,
                                  unsigned* __restrict__ amaxb,
                                  float* __restrict__ denom,
                                  int nc, int nh) {
    int i = blockIdx.x * 256 + threadIdx.x;
    if (i < nc) agg[i] = 0.f;
    if (i < nh) { amaxb[i] = 0u; denom[i] = 0.f; }
}

__global__ void residual_kernel(const float* __restrict__ agg,
                                const float* __restrict__ bias,
                                const float* __restrict__ xin,
                                float* __restrict__ xout, int n) {
    int i = blockIdx.x * 256 + threadIdx.x;
    if (i < n) xout[i] = agg[i] + bias[i % CH] + xin[i];
}

// ---------------------------------------------------------------------------
// WeE[l][d][h] = sum_c w_edge[l][d][c] * e[l][c][h]
// ---------------------------------------------------------------------------
__global__ __launch_bounds__(128)
void wee_kernel(const float* __restrict__ w_edge,
                const float* __restrict__ evec,
                float* __restrict__ WeE) {
    int ld = blockIdx.x;                 // l*EDIM + d
    int l = ld / EDIM, d = ld % EDIM;
    int t = threadIdx.x;
    const float* wrow = w_edge + ((size_t)l * EDIM + d) * CH;
    const float* eb   = evec + (size_t)l * CH * NH;
    float acc[NH] = {0.f, 0.f, 0.f, 0.f};
#pragma unroll
    for (int i = 0; i < 4; ++i) {
        int c = i * 128 + t;
        float wv = wrow[c];
#pragma unroll
        for (int h = 0; h < NH; ++h) acc[h] += wv * eb[c * NH + h];
    }
    __shared__ float sm[128 * NH];
#pragma unroll
    for (int h = 0; h < NH; ++h) sm[t * NH + h] = acc[h];
    __syncthreads();
    for (int off = 64; off > 0; off >>= 1) {
        if (t < off)
#pragma unroll
            for (int h = 0; h < NH; ++h)
                sm[t * NH + h] += sm[(t + off) * NH + h];
        __syncthreads();
    }
    if (t == 0)
#pragma unroll
        for (int h = 0; h < NH; ++h)
            WeE[((size_t)l * EDIM + d) * NH + h] = sm[h];
}

// ae[l][e][h] = sum_d edge_repre[e][d] * WeE[l][d][h]  (both layers, one pass)
__global__ __launch_bounds__(128)
void ae_kernel(const float* __restrict__ erep,
               const float* __restrict__ WeE,
               float* __restrict__ ae) {
    int e = blockIdx.x;
    int t = threadIdx.x;
    const float* er = erep + (size_t)e * EDIM;
    float acc[NLAY * NH] = {0.f,0.f,0.f,0.f,0.f,0.f,0.f,0.f};
#pragma unroll
    for (int i = 0; i < 6; ++i) {
        int d = i * 128 + t;
        float v = er[d];
#pragma unroll
        for (int l = 0; l < NLAY; ++l)
#pragma unroll
            for (int h = 0; h < NH; ++h)
                acc[l * NH + h] += v * WeE[((size_t)l * EDIM + d) * NH + h];
    }
    __shared__ float sm[128 * 8];
#pragma unroll
    for (int j = 0; j < 8; ++j) sm[t * 8 + j] = acc[j];
    __syncthreads();
    for (int off = 64; off > 0; off >>= 1) {
        if (t < off)
#pragma unroll
            for (int j = 0; j < 8; ++j)
                sm[t * 8 + j] += sm[(t + off) * 8 + j];
        __syncthreads();
    }
    if (t == 0)
#pragma unroll
        for (int l = 0; l < NLAY; ++l)
#pragma unroll
            for (int h = 0; h < NH; ++h)
                ae[((size_t)l * NEDGES + e) * NH + h] = sm[l * NH + h];
}

// xwq[r][n][h] = xw[r][n][:] . q[:][h]   (and same for k)
__global__ __launch_bounds__(128)
void proj_qk_kernel(const float* __restrict__ xw,
                    const float* __restrict__ q,
                    const float* __restrict__ k,
                    float* __restrict__ xwq,
                    float* __restrict__ xwk) {
    int n = blockIdx.x, r = blockIdx.y, t = threadIdx.x;
    const float* row = xw + ((size_t)r * NNODES + n) * CH;
    float qa[NH] = {0.f,0.f,0.f,0.f}, ka[NH] = {0.f,0.f,0.f,0.f};
#pragma unroll
    for (int i = 0; i < 4; ++i) {
        int c = i * 128 + t;
        float xv = row[c];
#pragma unroll
        for (int h = 0; h < NH; ++h) {
            qa[h] += xv * q[c * NH + h];
            ka[h] += xv * k[c * NH + h];
        }
    }
    __shared__ float sm[128 * 8];
#pragma unroll
    for (int h = 0; h < NH; ++h) { sm[t*8+h] = qa[h]; sm[t*8+4+h] = ka[h]; }
    __syncthreads();
    for (int off = 64; off > 0; off >>= 1) {
        if (t < off)
#pragma unroll
            for (int j = 0; j < 8; ++j)
                sm[t * 8 + j] += sm[(t + off) * 8 + j];
        __syncthreads();
    }
    if (t == 0) {
        size_t base = ((size_t)r * NNODES + n) * NH;
#pragma unroll
        for (int h = 0; h < NH; ++h) {
            xwq[base + h] = sm[h];
            xwk[base + h] = sm[4 + h];
        }
    }
}

// ---------------------------------------------------------------------------
// Segment softmax (monotonic uint key trick for float atomicMax)
// ---------------------------------------------------------------------------
__device__ inline unsigned fkey(float v) {
    unsigned u = __float_as_uint(v);
    return (u >> 31) ? ~u : (u | 0x80000000u);
}
__device__ inline float funkey(unsigned kk) {
    unsigned u = (kk >> 31) ? (kk & 0x7FFFFFFFu) : ~kk;
    return __uint_as_float(u);
}

__global__ void alpha_kernel(const int* __restrict__ src,
                             const int* __restrict__ dst,
                             const int* __restrict__ et,
                             const float* __restrict__ xwq,
                             const float* __restrict__ xwk,
                             const float* __restrict__ ae_l,
                             float* __restrict__ alpha,
                             unsigned* __restrict__ amaxb) {
    int e = blockIdx.x * 256 + threadIdx.x;
    if (e >= NEDGES) return;
    int s = src[e], d = dst[e], r = et[e];
    size_t bq = ((size_t)r * NNODES + d) * NH;
    size_t bk = ((size_t)r * NNODES + s) * NH;
#pragma unroll
    for (int h = 0; h < NH; ++h) {
        float a = xwq[bq + h] + xwk[bk + h] + ae_l[(size_t)e * NH + h];
        a = (a > 0.f) ? a : NEG_SLOPE * a;
        alpha[(size_t)e * NH + h] = a;
        atomicMax(&amaxb[d * NH + h], fkey(a));
    }
}

__global__ void ex_denom_kernel(const int* __restrict__ dst,
                                const unsigned* __restrict__ amaxb,
                                float* __restrict__ alpha,
                                float* __restrict__ denom) {
    int e = blockIdx.x * 256 + threadIdx.x;
    if (e >= NEDGES) return;
    int d = dst[e];
#pragma unroll
    for (int h = 0; h < NH; ++h) {
        float m = funkey(amaxb[d * NH + h]);
        float ex = __expf(alpha[(size_t)e * NH + h] - m);
        alpha[(size_t)e * NH + h] = ex;
        atomicAdd(&denom[d * NH + h], ex);
    }
}

// agg[dst] += (ex/denom) * xw[etype][src]   (one block per edge)
__global__ __launch_bounds__(128)
void message_kernel(const int* __restrict__ src,
                    const int* __restrict__ dst,
                    const int* __restrict__ et,
                    const float* __restrict__ ex,
                    const float* __restrict__ denom,
                    const float* __restrict__ xw,
                    float* __restrict__ agg) {
    int e = blockIdx.x;
    int t = threadIdx.x;
    int s = src[e], d = dst[e], r = et[e];
    int c0 = t * 4;
    int h  = c0 >> 7;  // 128 out-channels per head
    float coef = ex[(size_t)e * NH + h] / (denom[d * NH + h] + 1e-16f);
    const float* row = xw + ((size_t)r * NNODES + s) * CH;
    float4 v = *(const float4*)(row + c0);
    float* o = agg + (size_t)d * CH + c0;
    atomicAdd(o + 0, coef * v.x);
    atomicAdd(o + 1, coef * v.y);
    atomicAdd(o + 2, coef * v.z);
    atomicAdd(o + 3, coef * v.w);
}

// ---------------------------------------------------------------------------
// LayerNorm: v = in + pre; o = g*norm(v)+b + post; write f32 and/or bf16
// ---------------------------------------------------------------------------
__global__ __launch_bounds__(256)
void ln_kernel(const float* __restrict__ in,
               const float* __restrict__ pre,
               const float* __restrict__ post,
               const float* __restrict__ g,
               const float* __restrict__ b,
               float* __restrict__ outF,
               __bf16* __restrict__ outB) {
    int n = blockIdx.x, t = threadIdx.x;
    size_t base = (size_t)n * CH;
    float v0 = in[base + t]       + (pre ? pre[base + t]       : 0.f);
    float v1 = in[base + 256 + t] + (pre ? pre[base + 256 + t] : 0.f);
    __shared__ float ss[256], sq[256];
    ss[t] = v0 + v1;
    sq[t] = v0 * v0 + v1 * v1;
    __syncthreads();
    for (int off = 128; off > 0; off >>= 1) {
        if (t < off) { ss[t] += ss[t + off]; sq[t] += sq[t + off]; }
        __syncthreads();
    }
    float mean = ss[0] * (1.f / CH);
    float var  = sq[0] * (1.f / CH) - mean * mean;
    float inv  = rsqrtf(var + LN_EPS);
#pragma unroll
    for (int j = 0; j < 2; ++j) {
        int c = t + j * 256;
        float v = (j == 0 ? v0 : v1);
        float o = g[c] * (v - mean) * inv + b[c] + (post ? post[base + c] : 0.f);
        if (outF) outF[base + c] = o;
        if (outB) outB[base + c] = (__bf16)o;
    }
}

// ---------------------------------------------------------------------------
// Host-side orchestration
// ---------------------------------------------------------------------------
extern "C" void kernel_launch(void* const* d_in, const int* in_sizes, int n_in,
                              void* d_out, int out_size, void* d_ws, size_t ws_size,
                              hipStream_t stream) {
    const float* x_in  = (const float*)d_in[0];
    const int*   eidx  = (const int*)d_in[1];
    const int*   etyp  = (const int*)d_in[2];
    const float* erep  = (const float*)d_in[3];
    const float* Wts   = (const float*)d_in[4];
    const float* qw    = (const float*)d_in[5];
    const float* kw    = (const float*)d_in[6];
    const float* ew    = (const float*)d_in[7];
    const float* wedge = (const float*)d_in[8];
    const float* cbias = (const float*)d_in[9];
    const float* lng   = (const float*)d_in[10];
    const float* lnb   = (const float*)d_in[11];
    const float* f1w   = (const float*)d_in[12];
    const float* f1b   = (const float*)d_in[13];
    const float* f2w   = (const float*)d_in[14];
    const float* f2b   = (const float*)d_in[15];
    const int* srcp = eidx;
    const int* dstp = eidx + NEDGES;

    char* wsb = (char*)d_ws;
    size_t off = 0;
    auto take = [&](size_t bytes) -> char* {
        char* p = wsb + off;
        off = (off + bytes + 255) & ~(size_t)255;
        return p;
    };
    float*    xw    = (float*)   take((size_t)NREL * NNODES * CH * 4);   // 164 MB
    __bf16*   xbf   = (__bf16*)  take((size_t)NNODES * CH * 2);
    __bf16*   WbfT  = (__bf16*)  take((size_t)NREL * CH * CH * 2);
    float*    xwq   = (float*)   take((size_t)NREL * NNODES * NH * 4);
    float*    xwk   = (float*)   take((size_t)NREL * NNODES * NH * 4);
    float*    aeb   = (float*)   take((size_t)NLAY * NEDGES * NH * 4);
    float*    WeE   = (float*)   take((size_t)NLAY * EDIM * NH * 4);
    float*    alpha = (float*)   take((size_t)NEDGES * NH * 4);
    unsigned* amaxb = (unsigned*)take((size_t)NNODES * NH * 4);
    float*    denom = (float*)   take((size_t)NNODES * NH * 4);
    float*    agg   = (float*)   take((size_t)NNODES * CH * 4);
    float*    xcur  = (float*)   take((size_t)NNODES * CH * 4);
    float*    xln   = (float*)   take((size_t)NNODES * CH * 4);
    __bf16*   hid   = (__bf16*)  take((size_t)NNODES * HID * 2);
    __bf16*   ff1T  = (__bf16*)  take((size_t)CH * HID * 2);
    __bf16*   ff2T  = (__bf16*)  take((size_t)HID * CH * 2);

    const int NC = NNODES * CH;
    const dim3 blk256(256), blk128(128);
    const int gNC = (NC + 255) / 256;
    const int gE  = (NEDGES + 255) / 256;
    const int gMt = (NNODES + BM - 1) / BM;   // 79 row tiles

    // --- edge-attention precompute (both layers, one pass over edge_repre) ---
    wee_kernel<<<NLAY * EDIM, blk128, 0, stream>>>(wedge, ew, WeE);
    ae_kernel<<<NEDGES, blk128, 0, stream>>>(erep, WeE, aeb);

    // --- RGAT layers ---
    for (int l = 0; l < NLAY; ++l) {
        const float* xin   = (l == 0) ? x_in : xcur;
        const float* Wl    = Wts + (size_t)l * NREL * CH * CH;
        const float* ql    = qw + (size_t)l * CH * NH;
        const float* kl    = kw + (size_t)l * CH * NH;
        const float* ae_l  = aeb + (size_t)l * NEDGES * NH;
        const float* b_l   = cbias + (size_t)l * CH;

        f32_to_bf16_kernel<<<gNC, blk256, 0, stream>>>(xin, xbf, NC);
        transpose_bf16_kernel<<<(NREL * CH * CH + 255) / 256, blk256, 0, stream>>>(
            Wl, WbfT, NREL, CH, CH);

        // xw[r] = xbf @ W[r]  -> v_wmma_f32_16x16x32_bf16
        gemm_bf16_wmma<<<dim3(CH / BN, gMt, NREL), blk256, 0, stream>>>(
            xbf, WbfT, NNODES, CH, CH,
            (long)CH * CH, (long)NNODES * CH,
            nullptr, nullptr, xw, nullptr, 0);

        proj_qk_kernel<<<dim3(NNODES, NREL), blk128, 0, stream>>>(
            xw, ql, kl, xwq, xwk);

        init_layer_kernel<<<gNC, blk256, 0, stream>>>(
            agg, amaxb, denom, NC, NNODES * NH);
        alpha_kernel<<<gE, blk256, 0, stream>>>(
            srcp, dstp, etyp, xwq, xwk, ae_l, alpha, amaxb);
        ex_denom_kernel<<<gE, blk256, 0, stream>>>(dstp, amaxb, alpha, denom);
        message_kernel<<<NEDGES, blk128, 0, stream>>>(
            srcp, dstp, etyp, alpha, denom, xw, agg);

        residual_kernel<<<gNC, blk256, 0, stream>>>(agg, b_l, xin, xcur, NC);
    }

    // --- x = LN(x + enc_emb); also emit bf16 copy as FFN input ---
    ln_kernel<<<NNODES, blk256, 0, stream>>>(
        xcur, x_in, nullptr, lng, lnb, xln, xbf);

    // --- FFN (WMMA) ---
    transpose_bf16_kernel<<<(CH * HID + 255) / 256, blk256, 0, stream>>>(
        f1w, ff1T, 1, CH, HID);
    transpose_bf16_kernel<<<(HID * CH + 255) / 256, blk256, 0, stream>>>(
        f2w, ff2T, 1, HID, CH);

    // hid = relu(xln @ ff1 + b1), stored bf16
    gemm_bf16_wmma<<<dim3(HID / BN, gMt, 1), blk256, 0, stream>>>(
        xbf, ff1T, NNODES, CH, HID, 0, 0,
        f1b, nullptr, nullptr, hid, 1);
    // y = hid @ ff2 + b2 + xln   (reuse agg as y)
    gemm_bf16_wmma<<<dim3(CH / BN, gMt, 1), blk256, 0, stream>>>(
        hid, ff2T, NNODES, HID, CH, 0, 0,
        f2b, xln, agg, nullptr, 2);

    // --- out = LN(y) + enc_emb ---
    ln_kernel<<<NNODES, blk256, 0, stream>>>(
        agg, nullptr, x_in, lng, lnb, (float*)d_out, nullptr);
}
